// TopKSAE_78417512891016
// MI455X (gfx1250) — matrix operand order repaired and checked by reading, not compiled, not run on hardware
//
#include <hip/hip_runtime.h>

// ---- problem constants (compile-time; d_in[4]=k is assumed == 32) ----
#define BATCH 8192
#define DIN   1024
#define DEMB  16384
#define TOPK  32

typedef __attribute__((ext_vector_type(8)))  unsigned short u16x8;
typedef __attribute__((ext_vector_type(16))) unsigned short u16x16;
typedef __attribute__((ext_vector_type(8)))  float          v8f;
typedef __attribute__((ext_vector_type(16))) __bf16         v16bf;

// float -> bf16 bits, round-to-nearest-even
__device__ __forceinline__ unsigned short f2bf(float f) {
    union { float f; unsigned int u; } cv; cv.f = f;
    unsigned int u = cv.u;
    unsigned int r = u + 0x7fffu + ((u >> 16) & 1u);
    return (unsigned short)(r >> 16);
}

// combine two 8x-ushort LDS loads into one 16-wide bf16 fragment
__device__ __forceinline__ v16bf mk_frag(const unsigned short* p0, const unsigned short* p1) {
    u16x8 lo = *(const u16x8*)p0;
    u16x8 hi = *(const u16x8*)p1;
    u16x16 c;
#pragma unroll
    for (int i = 0; i < 8; ++i) { c[i] = lo[i]; c[i + 8] = hi[i]; }
    return __builtin_bit_cast(v16bf, c);
}

// LDS byte offset of a __shared__ object = low 32 bits of its flat address
__device__ __forceinline__ unsigned lds_off(const void* p) {
    return (unsigned)(unsigned long long)(size_t)p;
}

// async 16-byte copy global -> LDS (CDNA5, tracked by ASYNCcnt)
__device__ __forceinline__ void async_copy_b128(unsigned lds_byte_off,
                                                const unsigned short* gaddr) {
    asm volatile("global_load_async_to_lds_b128 %0, %1, off"
                 :: "v"(lds_byte_off), "v"(gaddr)
                 : "memory");
}

// ---------------------------------------------------------------------------
// K1: row-wise normalize (mean, unbiased std) and subtract decoder_b; emit bf16
// one block (256 threads = 8 waves) per row; 4 floats per thread
// ---------------------------------------------------------------------------
__global__ void sae_norm_kernel(const float* __restrict__ x,
                                const float* __restrict__ dec_b,
                                unsigned short* __restrict__ A) {
    const int row = blockIdx.x;
    const int t   = threadIdx.x;
    const float4 val = ((const float4*)(x + (size_t)row * DIN))[t];

    float s  = val.x + val.y + val.z + val.w;
    float sq = val.x * val.x + val.y * val.y + val.z * val.z + val.w * val.w;
#pragma unroll
    for (int o = 16; o > 0; o >>= 1) {
        s  += __shfl_xor(s,  o, 32);
        sq += __shfl_xor(sq, o, 32);
    }
    __shared__ float ws_s[8], ws_q[8];
    __shared__ float s_mean, s_rstd;
    const int wid = t >> 5, lane = t & 31;
    if (lane == 0) { ws_s[wid] = s; ws_q[wid] = sq; }
    __syncthreads();
    if (t == 0) {
        float S = 0.f, Q = 0.f;
#pragma unroll
        for (int i = 0; i < 8; ++i) { S += ws_s[i]; Q += ws_q[i]; }
        float mean = S / (float)DIN;
        float var  = (Q - (float)DIN * mean * mean) / (float)(DIN - 1);
        s_mean = mean;
        s_rstd = 1.0f / (sqrtf(var) + 1e-7f);
    }
    __syncthreads();
    const float mean = s_mean, rstd = s_rstd;

    const float4 db = ((const float4*)dec_b)[t];
    float y0 = (val.x - mean) * rstd - db.x;
    float y1 = (val.y - mean) * rstd - db.y;
    float y2 = (val.z - mean) * rstd - db.z;
    float y3 = (val.w - mean) * rstd - db.w;

    unsigned int lo = (unsigned int)f2bf(y0) | ((unsigned int)f2bf(y1) << 16);
    unsigned int hi = (unsigned int)f2bf(y2) | ((unsigned int)f2bf(y3) << 16);
    uint2* out = (uint2*)(A + (size_t)row * DIN + 4 * t);
    *out = make_uint2(lo, hi);
}

// ---------------------------------------------------------------------------
// K1b: transpose-convert encoder_w [DIN][DEMB] f32 -> Wt [DEMB][DIN] bf16
// 32x32 tiles through LDS, 256 threads
// ---------------------------------------------------------------------------
__global__ void sae_wt_kernel(const float* __restrict__ W,
                              unsigned short* __restrict__ Wt) {
    __shared__ unsigned short tile[32][33];
    const int n0 = blockIdx.x * 32;
    const int k0 = blockIdx.y * 32;
    const int tx = threadIdx.x & 31;
    const int ty = threadIdx.x >> 5;   // 0..7
#pragma unroll
    for (int r = ty; r < 32; r += 8)
        tile[r][tx] = f2bf(W[(size_t)(k0 + r) * DEMB + n0 + tx]);
    __syncthreads();
#pragma unroll
    for (int r = ty; r < 32; r += 8)
        Wt[(size_t)(n0 + r) * DIN + k0 + tx] = tile[tx][r];
}

// ---------------------------------------------------------------------------
// K2: GEMM e_pre = relu(A_bf16 @ Wt_bf16^T + enc_b) -> d_out (f32)
// block tile 128(M) x 128(N), BK=32; 8 waves, each wave = 16(M) x 128(N)
// Double-buffered LDS; tiles are fetched with global_load_async_to_lds_b128
// (ASYNCcnt) one full K-step ahead; fragments via paired ds_load_b128;
// 8x v_wmma_f32_16x16x32_bf16 per wave per K-step.
// ---------------------------------------------------------------------------
__global__ void sae_gemm_kernel(const unsigned short* __restrict__ A,
                                const unsigned short* __restrict__ Wt,
                                const float* __restrict__ enc_b,
                                float* __restrict__ out) {
    __shared__ unsigned short As[2][128 * 32];   // [m][k], k contiguous
    __shared__ unsigned short Bs[2][128 * 32];   // [n][k], k contiguous

    const int t    = threadIdx.x;
    const int w    = t >> 5;          // wave id 0..7
    const int lane = t & 31;
    const int half = lane >> 4;       // upper-K half select
    const int l15  = lane & 15;
    const int bm0  = blockIdx.y * 128;
    const int bn0  = blockIdx.x * 128;
    const int m    = w * 16 + l15;    // row within block tile (0..127)

    // this thread's two 16-byte segments per matrix: seg ids t and t+256
    const int r0 = t >> 2,         c0 = (t & 3) * 8;
    const int r1 = (t + 256) >> 2, c1 = ((t + 256) & 3) * 8;
    const unsigned short* gA0 = A  + (size_t)(bm0 + r0) * DIN + c0;
    const unsigned short* gA1 = A  + (size_t)(bm0 + r1) * DIN + c1;
    const unsigned short* gB0 = Wt + (size_t)(bn0 + r0) * DIN + c0;
    const unsigned short* gB1 = Wt + (size_t)(bn0 + r1) * DIN + c1;
    const unsigned lA0[2] = { lds_off(&As[0][r0 * 32 + c0]), lds_off(&As[1][r0 * 32 + c0]) };
    const unsigned lA1[2] = { lds_off(&As[0][r1 * 32 + c1]), lds_off(&As[1][r1 * 32 + c1]) };
    const unsigned lB0[2] = { lds_off(&Bs[0][r0 * 32 + c0]), lds_off(&Bs[1][r0 * 32 + c0]) };
    const unsigned lB1[2] = { lds_off(&Bs[0][r1 * 32 + c1]), lds_off(&Bs[1][r1 * 32 + c1]) };

    v8f acc[8];
#pragma unroll
    for (int j = 0; j < 8; ++j) {
        v8f z = {0.f, 0.f, 0.f, 0.f, 0.f, 0.f, 0.f, 0.f};
        acc[j] = z;
    }

    // prologue: prefetch K-tile 0 into buffer 0
    async_copy_b128(lA0[0], gA0);
    async_copy_b128(lA1[0], gA1);
    async_copy_b128(lB0[0], gB0);
    async_copy_b128(lB1[0], gB1);

    int buf = 0;
    for (int k0 = 0; k0 < DIN; k0 += 32) {
        const int nbuf = buf ^ 1;
        if (k0 + 32 < DIN) {
            // prefetch next K-tile into the other buffer
            async_copy_b128(lA0[nbuf], gA0 + k0 + 32);
            async_copy_b128(lA1[nbuf], gA1 + k0 + 32);
            async_copy_b128(lB0[nbuf], gB0 + k0 + 32);
            async_copy_b128(lB1[nbuf], gB1 + k0 + 32);
            // current tile's 4 async copies done; next tile's 4 still in flight
            asm volatile("s_wait_asynccnt 0x4" ::: "memory");
        } else {
            asm volatile("s_wait_asynccnt 0x0" ::: "memory");
        }
        __syncthreads();

        // A fragment: lane holds M=l15; VGPR0-3 -> K=half*8.., VGPR4-7 -> K=half*8+16..
        const unsigned short* as = &As[buf][m * 32 + half * 8];
        const v16bf a = mk_frag(as, as + 16);

#pragma unroll
        for (int j = 0; j < 8; ++j) {
            const int n = j * 16 + l15;
            // B fragment: lane holds N=l15; VGPR0-3 -> K=half*8.., VGPR4-7 -> K=half*8+16..
            const unsigned short* bs = &Bs[buf][n * 32 + half * 8];
            const v16bf b = mk_frag(bs, bs + 16);
            acc[j] = __builtin_amdgcn_wmma_f32_16x16x32_bf16(
                /*neg_a=*/false, a, /*neg_b=*/false, b,
                /*c_mod=*/(short)0, acc[j], /*reuse_a=*/false, /*reuse_b=*/false);
        }
        __syncthreads();   // all waves done reading buf before it is refilled
        buf = nbuf;
    }

    // epilogue: bias + relu, C layout: (VGPR r, lane) -> row = r + half*8, col = l15
#pragma unroll
    for (int j = 0; j < 8; ++j) {
        const int col = bn0 + j * 16 + l15;
        const float bias = enc_b[col];
#pragma unroll
        for (int r = 0; r < 8; ++r) {
            const int row = bm0 + w * 16 + half * 8 + r;
            float vv = fmaxf(acc[j][r] + bias, 0.0f);
            out[(size_t)row * DEMB + col] = vv;
        }
    }
}

// ---------------------------------------------------------------------------
// K3: in-place top-32 sparsification of each row of d_out.
// One block per row; 64 values/thread in registers (static indexing only).
// 32 tournament rounds with lazy local-argmax recompute; winners stay in
// memory untouched, everything else is overwritten with zeros.
// ---------------------------------------------------------------------------
__global__ void sae_topk_kernel(float* __restrict__ out) {
    const int row = blockIdx.x;
    const int t   = threadIdx.x;
    float* rp = out + (size_t)row * DEMB;

    // thread owns elements idx = g*1024 + 4*t + c, g in [0,16), c in [0,4)
    float v[64];
#pragma unroll
    for (int g = 0; g < 16; ++g) {
        float4 ld = *(const float4*)&rp[g * 1024 + 4 * t];
        v[g * 4 + 0] = ld.x; v[g * 4 + 1] = ld.y;
        v[g * 4 + 2] = ld.z; v[g * 4 + 3] = ld.w;
    }

    float lm = -1.0f; int li = 0;     // relu output >= 0, so -1 is a safe sentinel
#pragma unroll
    for (int s = 0; s < 64; ++s) {
        const int idx = (s >> 2) * 1024 + 4 * t + (s & 3);
        if (v[s] > lm) { lm = v[s]; li = idx; }
    }

    __shared__ float swv[8];
    __shared__ int   swi[8];
    __shared__ float bwv;
    __shared__ int   bwi;
    unsigned long long winmask = 0ull;

    for (int r = 0; r < TOPK; ++r) {
        float mv = lm; int mi = li;
#pragma unroll
        for (int o = 16; o > 0; o >>= 1) {
            float ov = __shfl_xor(mv, o, 32);
            int   oi = __shfl_xor(mi, o, 32);
            if (ov > mv || (ov == mv && oi < mi)) { mv = ov; mi = oi; }
        }
        if ((t & 31) == 0) { swv[t >> 5] = mv; swi[t >> 5] = mi; }
        __syncthreads();
        if (t == 0) {
            float bv = swv[0]; int bi = swi[0];
#pragma unroll
            for (int i = 1; i < 8; ++i)
                if (swv[i] > bv || (swv[i] == bv && swi[i] < bi)) { bv = swv[i]; bi = swi[i]; }
            bwv = bv; bwi = bi;
        }
        __syncthreads();
        const int widx = bwi;
        if (((widx >> 2) & 255) == t) {      // this thread owns the winner
            const int slot = ((widx >> 10) << 2) | (widx & 3);
            winmask |= (1ull << slot);
            lm = -1.0f; li = 0;
#pragma unroll
            for (int s = 0; s < 64; ++s) {
                if (s == slot) v[s] = -1.0f;
                const int idx = (s >> 2) * 1024 + 4 * t + (s & 3);
                if (v[s] > lm) { lm = v[s]; li = idx; }
            }
        }
        __syncthreads();
    }

    // zero everything that is not a winner; winners keep their e_pre value
#pragma unroll
    for (int g = 0; g < 16; ++g) {
        const unsigned gm = (unsigned)((winmask >> (g * 4)) & 0xFull);
        float* p = &rp[g * 1024 + 4 * t];
        if (gm == 0u) {
            *(float4*)p = make_float4(0.f, 0.f, 0.f, 0.f);
        } else {
#pragma unroll
            for (int c = 0; c < 4; ++c)
                if (!((gm >> c) & 1u)) p[c] = 0.f;
        }
    }
}

// ---------------------------------------------------------------------------
extern "C" void kernel_launch(void* const* d_in, const int* in_sizes, int n_in,
                              void* d_out, int out_size, void* d_ws, size_t ws_size,
                              hipStream_t stream) {
    const float* x     = (const float*)d_in[0];   // [8192,1024]
    const float* enc_w = (const float*)d_in[1];   // [1024,16384]
    const float* enc_b = (const float*)d_in[2];   // [16384]
    const float* dec_b = (const float*)d_in[3];   // [1024]
    // d_in[4] = k (assumed 32)

    unsigned short* Abf = (unsigned short*)d_ws;                                    // 16 MB
    unsigned short* Wt  = (unsigned short*)((char*)d_ws + (size_t)BATCH * DIN * 2); // 32 MB
    float* out = (float*)d_out;

    sae_norm_kernel<<<BATCH, 256, 0, stream>>>(x, dec_b, Abf);
    sae_wt_kernel<<<dim3(DEMB / 32, DIN / 32), 256, 0, stream>>>(enc_w, Wt);
    sae_gemm_kernel<<<dim3(DEMB / 128, BATCH / 128), 256, 0, stream>>>(Abf, Wt, enc_b, out);
    sae_topk_kernel<<<BATCH, 256, 0, stream>>>(out);
}
